// BiRNN_IFBUS3S8SAPP_84636625535343
// MI455X (gfx1250) — matrix operand order repaired and checked by reading, not compiled, not run on hardware
//
#include <hip/hip_runtime.h>
#include <hip/hip_bf16.h>

typedef unsigned short u16;
typedef __attribute__((ext_vector_type(16))) __bf16 v16bf;
typedef __attribute__((ext_vector_type(8)))  float  v8f;

#define B_  32
#define T_  2048
#define F_  64
#define U_  256
#define G_  1024   // 4*U
#define H_  512    // 2*U
#define NWG_DIR 8  // workgroups per direction; each owns 128 gate cols / 32 h cols

__device__ __forceinline__ u16 f2bf(float f) {
    unsigned u = __float_as_uint(f);
    unsigned r = u + 0x7FFFu + ((u >> 16) & 1u);
    return (u16)(r >> 16);
}
__device__ __forceinline__ float bf2f(u16 h) {
    return __uint_as_float(((unsigned)h) << 16);
}

// ---------------- prep kernels ----------------

// x (B,T,F) f32 -> time-major bf16 [T][B][F]
__global__ void prep_x_kernel(const float* __restrict__ x, u16* __restrict__ xo) {
    int idx = blockIdx.x * 256 + threadIdx.x;
    if (idx >= B_ * T_ * F_) return;
    int b = idx / (T_ * F_);
    int r = idx - b * (T_ * F_);
    int t = r / F_;
    int f = r - t * F_;
    xo[((size_t)t * B_ + b) * F_ + f] = f2bf(x[idx]);
}

// pack mask bits: bit b of mb[t] = (x_mask[b][t] != 1)
__global__ void prep_mask_kernel(const int* __restrict__ xm, unsigned* __restrict__ mb) {
    int t = blockIdx.x * 256 + threadIdx.x;
    if (t >= T_) return;
    unsigned m = 0;
    for (int b = 0; b < B_; ++b)
        if (xm[(size_t)b * T_ + t] != 1) m |= (1u << b);
    mb[t] = m;
}

// Swizzle [W;U] (K x 1024, K=F_in+256) f32 into WMMA-B fragment order, bf16:
// out[nt][kt][lane][e]; col=nt*16+(lane&15); h=lane>>4: e<8 -> k=kt*32+8h+e; e>=8 -> k=kt*32+16+8h+(e-8)
__global__ void prep_wgt_kernel(const float* __restrict__ W, const float* __restrict__ U,
                                int F_in, int Kt, u16* __restrict__ out) {
    int idx = blockIdx.x * 256 + threadIdx.x;
    int total = 64 * Kt * 512;
    if (idx >= total) return;
    int e    = idx & 15;
    int lane = (idx >> 4) & 31;
    int kt   = (idx >> 9) % Kt;
    int nt   = idx / (512 * Kt);
    int col  = nt * 16 + (lane & 15);
    int k    = kt * 32 + ((e >> 3) << 4) + ((lane >> 4) << 3) + (e & 7);
    float v  = (k < F_in) ? W[(size_t)k * G_ + col] : U[(size_t)(k - F_in) * G_ + col];
    out[idx] = f2bf(v);
}

__global__ void init_kernel(u16* __restrict__ hexch, unsigned* __restrict__ bars) {
    int tid = threadIdx.x;
    unsigned* h32 = (unsigned*)hexch;                // 4 * 32 * 256 u16 = 16384 u32
    for (int i = tid; i < 16384; i += 256) h32[i] = 0u;
    if (tid < 8) bars[tid] = 0u;
}

// ---- async global->LDS copy of one x tile slice (per wave: PX b128 DMAs) ----
template<int PX>
__device__ __forceinline__ void async_copy_x(const u16* __restrict__ src,
                                             const char* smembase,
                                             u16* dst, int w, int lane) {
    unsigned ldsbase = (unsigned)(size_t)((const char*)dst - smembase);
#pragma unroll
    for (int q = 0; q < PX; ++q) {
        unsigned off = (unsigned)((w * PX + q) * 512 + lane * 16);
        unsigned lds = ldsbase + off;
        unsigned long long ga = (unsigned long long)(size_t)((const char*)src + off);
        asm volatile("global_load_async_to_lds_b128 %0, %1, off"
                     :: "v"(lds), "v"(ga) : "memory");
    }
}

// ---------------- persistent bi-LSTM layer kernel ----------------
// grid = 16 blocks (dir = blk>>3, g = blk&7), 256 threads (8 wave32).
// Each WG owns h columns [g*32, g*32+32): wave w works on global ntile
// nt = (w>>1)*16 + 2*g + (w&1)  (gate block i/f/g/o x 2 subtiles).
// Weight B-fragments are register-resident for the whole 2048-step scan;
// x_t tiles are double-buffered in LDS via async global->LDS DMA.
template<int F_IN, int KT>
__global__ __launch_bounds__(256, 1)
void lstm_layer_kernel(const u16* __restrict__ xin,        // [T][32][F_IN] bf16
                       const unsigned* __restrict__ maskbits,
                       const u16* __restrict__ wgt,        // [2][64][KT][32][16] bf16
                       const float* __restrict__ bias_f,
                       const float* __restrict__ bias_b,
                       u16* __restrict__ hexch,            // [2][32*256] bf16
                       unsigned* __restrict__ bars,        // [2]
                       u16* __restrict__ hout) {           // [T][32][512] bf16
    constexpr int KX = F_IN / 32;          // k-tiles sourced from x
    constexpr int KH = U_ / 32;            // k-tiles sourced from h (8)
    static_assert(KX + KH == KT, "K tiling mismatch");
    constexpr int XBYTES = 32 * F_IN * 2;  // one x tile
    constexpr int PX = XBYTES / (512 * 8); // async b128 per wave (L0:1, L1:8)

    const int tid  = threadIdx.x;
    const int w    = tid >> 5;      // wave 0..7
    const int lane = tid & 31;
    const int dir  = blockIdx.x >> 3;
    const int g    = blockIdx.x & 7;

    extern __shared__ char smem[];
    u16*   Xs   = (u16*)smem;                       // [2][32][F_IN] double-buffered x
    u16*   Hs   = Xs + 2 * 32 * F_IN;               // [32][256] h_prev staging
    float* zb   = (float*)(Hs + 32 * U_);           // 8 waves x 32 rows x 16 cols
    float* cbuf = zb + 4096;                        // 32 x 32 f32 cell state slice
    float* hbuf = cbuf + 1024;                      // 32 x 32 f32 hidden slice
    float* blds = hbuf + 1024;                      // 128 biases (i|f|g|o)

    // ---- weights: load this wave's ntile B-fragments into registers (stay for all T) ----
    v16bf wreg[KT];
    {
        int nt = ((w >> 1) << 4) + 2 * g + (w & 1);
        const u16* wp = wgt + (size_t)dir * 64 * KT * 512
                            + (size_t)nt * KT * 512 + (size_t)lane * 16;
#pragma unroll
        for (int kt = 0; kt < KT; ++kt)
            wreg[kt] = *(const v16bf*)(wp + (size_t)kt * 512);
    }

    for (int i = tid; i < 1024; i += 256) { cbuf[i] = 0.f; hbuf[i] = 0.f; }
    if (tid < 128) {
        int blk = tid >> 5, j = tid & 31;
        const float* bsrc = dir ? bias_b : bias_f;
        blds[tid] = bsrc[blk * U_ + g * 32 + j];
    }

    u16* hex = hexch + (size_t)dir * (B_ * U_);
    const int half = lane >> 4, rr15 = lane & 15;
    const int t0 = dir ? (T_ - 1) : 0;

    // pre-issue async copy of the first x tile into buffer 0
    async_copy_x<PX>(xin + (size_t)t0 * B_ * F_IN, smem, Xs, w, lane);

    for (int s = 0; s < T_; ++s) {
        const int t = dir ? (T_ - 1 - s) : s;

        // wait for this step's x tile DMA (own slice; barrier below covers all slices)
        asm volatile("s_wait_asynccnt 0" ::: "memory");

        // stage h_prev (32x256 bf16) from global exchange buffer into LDS
        {
            const uint4* hs = (const uint4*)hex;
            for (int i = tid; i < 1024; i += 256) {
                int row = i >> 5, c8 = i & 31;
                *(uint4*)(Hs + (size_t)row * U_ + c8 * 8) = hs[(size_t)row * 32 + c8];
            }
        }
        __syncthreads();

        // kick off async DMA of the next x tile into the other buffer (overlaps WMMA)
        {
            int sn = (s + 1 < T_) ? (s + 1) : s;
            int tn = dir ? (T_ - 1 - sn) : sn;
            async_copy_x<PX>(xin + (size_t)tn * B_ * F_IN, smem,
                             Xs + ((s + 1) & 1) * 32 * F_IN, w, lane);
        }

        // ---- z = [x_t | h] x B via WMMA bf16 (B from registers) ----
        const u16* Xb = Xs + (s & 1) * 32 * F_IN;
        v8f acc0 = {0.f,0.f,0.f,0.f,0.f,0.f,0.f,0.f};
        v8f acc1 = {0.f,0.f,0.f,0.f,0.f,0.f,0.f,0.f};
#pragma unroll
        for (int kt = 0; kt < KX; ++kt) {
            const u16* a0p = Xb + (size_t)rr15 * F_IN + kt * 32 + half * 8;
            const u16* a1p = a0p + (size_t)16 * F_IN;
            v16bf a0, a1;
            ((uint4*)&a0)[0] = *(const uint4*)a0p;
            ((uint4*)&a0)[1] = *(const uint4*)(a0p + 16);
            ((uint4*)&a1)[0] = *(const uint4*)a1p;
            ((uint4*)&a1)[1] = *(const uint4*)(a1p + 16);
            acc0 = __builtin_amdgcn_wmma_f32_16x16x32_bf16(false, a0, false, wreg[kt],
                                                           (short)0, acc0, false, false);
            acc1 = __builtin_amdgcn_wmma_f32_16x16x32_bf16(false, a1, false, wreg[kt],
                                                           (short)0, acc1, false, false);
        }
#pragma unroll
        for (int kt = 0; kt < KH; ++kt) {
            const u16* a0p = Hs + (size_t)rr15 * U_ + kt * 32 + half * 8;
            const u16* a1p = a0p + (size_t)16 * U_;
            v16bf a0, a1;
            ((uint4*)&a0)[0] = *(const uint4*)a0p;
            ((uint4*)&a0)[1] = *(const uint4*)(a0p + 16);
            ((uint4*)&a1)[0] = *(const uint4*)a1p;
            ((uint4*)&a1)[1] = *(const uint4*)(a1p + 16);
            acc0 = __builtin_amdgcn_wmma_f32_16x16x32_bf16(false, a0, false, wreg[KX + kt],
                                                           (short)0, acc0, false, false);
            acc1 = __builtin_amdgcn_wmma_f32_16x16x32_bf16(false, a1, false, wreg[KX + kt],
                                                           (short)0, acc1, false, false);
        }
        // C layout: VGPR r, lane l -> row=(l>>4)*8+r (per M-tile), col=l&15
        {
            float* zw = zb + w * 512;
#pragma unroll
            for (int r = 0; r < 8; ++r) {
                zw[(half * 8 + r) * 16 + rr15]      = acc0[r];
                zw[(16 + half * 8 + r) * 16 + rr15] = acc1[r];
            }
        }
        __syncthreads();

        // ---- gates + state update for our 32 h-columns x 32 batch ----
        const unsigned mb = maskbits[t];
#pragma unroll
        for (int kk = 0; kk < 4; ++kk) {
            int e = tid + kk * 256;
            int b = e >> 5, j = e & 31;
            int sub = j >> 4, col = j & 15;
            float zi = zb[(0 + sub) * 512 + b * 16 + col] + blds[j];
            float zf = zb[(2 + sub) * 512 + b * 16 + col] + blds[32 + j];
            float zg = zb[(4 + sub) * 512 + b * 16 + col] + blds[64 + j];
            float zo = zb[(6 + sub) * 512 + b * 16 + col] + blds[96 + j];
            float ig = 1.f / (1.f + __expf(-zi));
            float fg = 1.f / (1.f + __expf(-zf));
            float gg = tanhf(zg);
            float og = 1.f / (1.f + __expf(-zo));
            float cp = cbuf[e], hp = hbuf[e];
            float cn = fg * cp + ig * gg;
            float hn = og * tanhf(cn);
            if (!((mb >> b) & 1u)) { cn = cp; hn = hp; }
            cbuf[e] = cn; hbuf[e] = hn;
            u16 hb = f2bf(hn);
            hex[(size_t)b * U_ + g * 32 + j] = hb;
            hout[((size_t)t * B_ + b) * H_ + dir * U_ + g * 32 + j] = hb;
        }
        __threadfence();
        __syncthreads();

        // ---- inter-WG step barrier (per direction), monotonic counter ----
        if (tid == 0) {
            atomicAdd(&bars[dir], 1u);
            unsigned target = (unsigned)NWG_DIR * (unsigned)(s + 1);
            while (__hip_atomic_load(&bars[dir], __ATOMIC_ACQUIRE,
                                     __HIP_MEMORY_SCOPE_AGENT) < target) {
                __builtin_amdgcn_s_sleep(2);
            }
        }
        __syncthreads();
    }
}

// ---------------- output heads ----------------
__global__ __launch_bounds__(256)
void heads_kernel(const u16* __restrict__ h1,   // [T][32][512] bf16 (pre-ReLU)
                  const float* __restrict__ Wp,  const float* __restrict__ bp,
                  const float* __restrict__ Wbu, const float* __restrict__ bbu,
                  const float* __restrict__ W3,  const float* __restrict__ b3,
                  const float* __restrict__ W8,  const float* __restrict__ b8,
                  const float* __restrict__ Wa,  const float* __restrict__ ba,
                  const float* __restrict__ Wpp, const float* __restrict__ bpp,
                  float* __restrict__ out) {
    __shared__ float wl[H_ * 20];
    __shared__ float bl[20];
    const int tid = threadIdx.x;
    for (int idx = tid; idx < H_ * 20; idx += 256) {
        int h = idx / 20, c = idx - h * 20;
        float v;
        if      (c < 2)  v = Wp [(size_t)h * 2 + c];
        else if (c < 4)  v = Wbu[(size_t)h * 2 + (c - 2)];
        else if (c < 7)  v = W3 [(size_t)h * 3 + (c - 4)];
        else if (c < 15) v = W8 [(size_t)h * 8 + (c - 7)];
        else if (c < 16) v = Wa [h];
        else             v = Wpp[(size_t)h * 4 + (c - 16)];
        wl[idx] = v;
    }
    if (tid < 20) {
        int c = tid;
        bl[c] = (c < 2)  ? bp[c]        : (c < 4)  ? bbu[c - 2]
              : (c < 7)  ? b3[c - 4]    : (c < 15) ? b8[c - 7]
              : (c < 16) ? ba[0]        : bpp[c - 16];
    }
    __syncthreads();

    const int n = blockIdx.x * 256 + tid;           // n = b*T + t
    const int b = n >> 11, t = n & (T_ - 1);
    const u16* row = h1 + ((size_t)t * B_ + b) * H_;

    float acc[20];
#pragma unroll
    for (int c = 0; c < 20; ++c) acc[c] = bl[c];
    for (int k = 0; k < H_; ++k) {
        float x = bf2f(row[k]);
        x = x > 0.f ? x : 0.f;                      // ReLU
        const float* wr = &wl[k * 20];
#pragma unroll
        for (int c = 0; c < 20; ++c) acc[c] += x * wr[c];
    }
    const size_t BT = (size_t)B_ * T_;
    // ppi = softmax(acc[0..1])
    {
        float m = fmaxf(acc[0], acc[1]);
        float e0 = __expf(acc[0] - m), e1 = __expf(acc[1] - m);
        float inv = 1.f / (e0 + e1);
        out[(size_t)n * 2 + 0] = e0 * inv;
        out[(size_t)n * 2 + 1] = e1 * inv;
    }
    out[BT * 2 + (size_t)n * 2 + 0] = acc[2];
    out[BT * 2 + (size_t)n * 2 + 1] = acc[3];
    for (int c = 0; c < 3; ++c) out[BT * 4  + (size_t)n * 3 + c] = acc[4 + c];
    for (int c = 0; c < 8; ++c) out[BT * 7  + (size_t)n * 8 + c] = acc[7 + c];
    out[BT * 15 + (size_t)n] = acc[15];
    for (int c = 0; c < 4; ++c) out[BT * 16 + (size_t)n * 4 + c] = acc[16 + c];
}

// ---------------- host ----------------
static inline size_t alignup(size_t x) { return (x + 255) & ~(size_t)255; }

extern "C" void kernel_launch(void* const* d_in, const int* in_sizes, int n_in,
                              void* d_out, int out_size, void* d_ws, size_t ws_size,
                              hipStream_t stream) {
    (void)in_sizes; (void)n_in; (void)out_size; (void)ws_size;
    const float* x      = (const float*)d_in[0];
    const int*   x_mask = (const int*)d_in[1];
    const float* Wf0 = (const float*)d_in[3];
    const float* Uf0 = (const float*)d_in[4];
    const float* bf0 = (const float*)d_in[5];
    const float* Wb0 = (const float*)d_in[6];
    const float* Ub0 = (const float*)d_in[7];
    const float* bb0 = (const float*)d_in[8];
    const float* Wf1 = (const float*)d_in[9];
    const float* Uf1 = (const float*)d_in[10];
    const float* bf1 = (const float*)d_in[11];
    const float* Wb1 = (const float*)d_in[12];
    const float* Ub1 = (const float*)d_in[13];
    const float* bb1 = (const float*)d_in[14];
    const float* Wp  = (const float*)d_in[15]; const float* bp  = (const float*)d_in[16];
    const float* Wbu = (const float*)d_in[17]; const float* bbu = (const float*)d_in[18];
    const float* W3  = (const float*)d_in[19]; const float* b3  = (const float*)d_in[20];
    const float* W8  = (const float*)d_in[21]; const float* b8  = (const float*)d_in[22];
    const float* Wa  = (const float*)d_in[23]; const float* ba  = (const float*)d_in[24];
    const float* Wpp = (const float*)d_in[25]; const float* bpp = (const float*)d_in[26];
    float* out = (float*)d_out;

    char* ws = (char*)d_ws;
    size_t o = 0;
    const int Kt0 = (F_ + U_) / 32;   // 10
    const int Kt1 = (H_ + U_) / 32;   // 24
    u16* xin0   = (u16*)(ws + o); o = alignup(o + (size_t)T_ * B_ * F_ * 2);
    u16* h0     = (u16*)(ws + o); o = alignup(o + (size_t)T_ * B_ * H_ * 2);
    u16* h1     = (u16*)(ws + o); o = alignup(o + (size_t)T_ * B_ * H_ * 2);
    u16* wgt0   = (u16*)(ws + o); o = alignup(o + (size_t)2 * 64 * Kt0 * 512 * 2);
    u16* wgt1   = (u16*)(ws + o); o = alignup(o + (size_t)2 * 64 * Kt1 * 512 * 2);
    unsigned* maskbits = (unsigned*)(ws + o); o = alignup(o + (size_t)T_ * 4);
    u16* hexch  = (u16*)(ws + o); o = alignup(o + (size_t)4 * B_ * U_ * 2);
    unsigned* bars = (unsigned*)(ws + o); o = alignup(o + 8 * sizeof(unsigned));

    init_kernel<<<1, 256, 0, stream>>>(hexch, bars);
    prep_x_kernel<<<(B_ * T_ * F_ + 255) / 256, 256, 0, stream>>>(x, xin0);
    prep_mask_kernel<<<(T_ + 255) / 256, 256, 0, stream>>>(x_mask, maskbits);

    int wtot0 = 64 * Kt0 * 512, wtot1 = 64 * Kt1 * 512;
    prep_wgt_kernel<<<(wtot0 + 255) / 256, 256, 0, stream>>>(Wf0, Uf0, F_, Kt0, wgt0);
    prep_wgt_kernel<<<(wtot0 + 255) / 256, 256, 0, stream>>>(Wb0, Ub0, F_, Kt0, wgt0 + (size_t)wtot0);
    prep_wgt_kernel<<<(wtot1 + 255) / 256, 256, 0, stream>>>(Wf1, Uf1, H_, Kt1, wgt1);
    prep_wgt_kernel<<<(wtot1 + 255) / 256, 256, 0, stream>>>(Wb1, Ub1, H_, Kt1, wgt1 + (size_t)wtot1);

    auto ldsBytes = [](int F_in) -> size_t {
        return (size_t)2 * 32 * F_in * 2          // x double buffer
             + (size_t)32 * U_ * 2                // h staging
             + 4096 * 4 + 1024 * 4 + 1024 * 4 + 128 * 4;
    };
    lstm_layer_kernel<F_, 10><<<16, 256, ldsBytes(F_), stream>>>(
        xin0, maskbits, wgt0, bf0, bb0, hexch, bars, h0);
    lstm_layer_kernel<H_, 24><<<16, 256, ldsBytes(H_), stream>>>(
        h0, maskbits, wgt1, bf1, bb1, hexch + 2 * B_ * U_, bars + 2, h1);

    heads_kernel<<<(B_ * T_) / 256, 256, 0, stream>>>(
        h1, Wp, bp, Wbu, bbu, W3, b3, W8, b8, Wa, ba, Wpp, bpp, out);
}
